// MultiHeadDotProductAttention_11149735100894
// MI455X (gfx1250) — matrix operand order repaired and checked by reading, not compiled, not run on hardware
//
#include <hip/hip_runtime.h>

#define DEVINL __device__ __forceinline__

typedef __attribute__((ext_vector_type(16))) __bf16 v16bf;
typedef __attribute__((ext_vector_type(8)))  __bf16 bf16x8;
typedef __attribute__((ext_vector_type(8)))  float  v8f;
typedef __attribute__((ext_vector_type(4)))  int    v4i;

static constexpr int B_ = 16, L_ = 1024, F_ = 512, H_ = 8, D_ = 64;
static constexpr int HD = H_ * D_; // 512

// ---------------------------------------------------------------------------
// CDNA5 async global->LDS copy (ASYNCcnt path), with safe fallback.
// Builtin signature on this toolchain takes v4i pointees (per round-2 diag):
//   (addrspace(1) v4i*, addrspace(3) v4i*, imm offset, imm cpol)
// ---------------------------------------------------------------------------
#if __has_builtin(__builtin_amdgcn_global_load_async_to_lds_b128)
#define USE_ASYNC_LDS 1
#else
#define USE_ASYNC_LDS 0
#endif

typedef __attribute__((address_space(1))) v4i global_v4i;
typedef __attribute__((address_space(3))) v4i lds_v4i;

DEVINL void async_cp16(const __bf16* g, __bf16* l) {
#if USE_ASYNC_LDS
  __builtin_amdgcn_global_load_async_to_lds_b128(
      (global_v4i*)g, (lds_v4i*)l, 0, 0);
#else
  *(bf16x8*)l = *(const bf16x8*)g;
#endif
}

DEVINL void wait_async() {
#if USE_ASYNC_LDS
#if __has_builtin(__builtin_amdgcn_s_wait_asynccnt)
  __builtin_amdgcn_s_wait_asynccnt(0);
#else
  asm volatile("s_wait_asynccnt 0" ::: "memory");
#endif
#endif
}

// Load a 16x(8+8) bf16 fragment: two 16-byte chunks at p and p+gap.
DEVINL v16bf frag_ld2(const __bf16* p, int gap) {
  bf16x8 lo = *(const bf16x8*)(p);
  bf16x8 hi = *(const bf16x8*)(p + gap);
  v16bf r;
#pragma unroll
  for (int i = 0; i < 8; ++i) { r[i] = lo[i]; r[i + 8] = hi[i]; }
  return r;
}

// A fragment: 16(M) x 32(K) from a row-major tile (stride in elements).
// Lanes 0-15: row=lane, K {0..7,16..23}; lanes 16-31: row=lane-16, K {8..15,24..31}.
DEVINL v16bf frag_a(const __bf16* tile, int stride, int rowBase, int colBase, unsigned lane) {
  int r = rowBase + (int)(lane & 15u);
  int koff = (lane >> 4) ? 8 : 0;
  return frag_ld2(tile + (size_t)r * stride + colBase + koff, 16);
}

// B fragment: 32(K) x 16(N) sourced from the *transposed* (N x K row-major) tile.
// Lanes 0-15: col n=lane, K {0..7,16..23}; lanes 16-31: n=lane-16, K {8..15,24..31}.
DEVINL v16bf frag_bt(const __bf16* tileT, int stride, int rowBase, int colBase, unsigned lane) {
  int n = rowBase + (int)(lane & 15u);
  int koff = (lane >> 4) ? 8 : 0;
  return frag_ld2(tileT + (size_t)n * stride + colBase + koff, 16);
}

DEVINL v8f wmma_bf16(v16bf a, v16bf b, v8f c) {
  return __builtin_amdgcn_wmma_f32_16x16x32_bf16(false, a, false, b, (short)0, c, false, false);
}

// ---------------------------------------------------------------------------
// Kernel 1: fused Q/K/V projection GEMM. grid=(M/256, 512/64, 3), block=256.
// Each wave computes a 32x64 strip (8 WMMA per 32-K step, B frags reused 2x).
// Q,K stored [b][h][l][d] bf16; V stored transposed [b][h][d][l] bf16.
// ---------------------------------------------------------------------------
__global__ __launch_bounds__(256)
void qkv_gemm(const float* __restrict__ inq, const float* __restrict__ inkv,
              const float* __restrict__ Wq, const float* __restrict__ bq,
              const float* __restrict__ Wk, const float* __restrict__ bk,
              const float* __restrict__ Wv, const float* __restrict__ bv,
              __bf16* __restrict__ Qo, __bf16* __restrict__ Ko, __bf16* __restrict__ Vo) {
  __shared__ __bf16 As[256 * 40];  // 256 rows x 32 K, stride 40 (80B, conflict-free)
  __shared__ __bf16 Bt[64 * 40];   // transposed: 64 N x 32 K

  const int t = threadIdx.x;
  const unsigned lane = t & 31u;
  const int wave = t >> 5;
  const int mBase = blockIdx.x * 256;
  const int nBase = blockIdx.y * 64;
  const int z = blockIdx.z;

  const float* A = (z == 0) ? inq : inkv;
  const float* W = (z == 0) ? Wq : (z == 1) ? Wk : Wv;
  const float* bias = (z == 0) ? bq : (z == 1) ? bk : bv;

  v8f acc[2][4];
#pragma unroll
  for (int mt = 0; mt < 2; ++mt)
#pragma unroll
    for (int nt = 0; nt < 4; ++nt)
#pragma unroll
      for (int j = 0; j < 8; ++j) acc[mt][nt][j] = 0.f;

  for (int k0 = 0; k0 < F_; k0 += 32) {
    __syncthreads();
    // Stage A: 256x32 f32 -> bf16 (2048 float4 chunks, 8 per thread)
#pragma unroll
    for (int i = 0; i < 8; ++i) {
      int idx = i * 256 + t;
      int row = idx >> 3;
      int c4 = (idx & 7) * 4;
      const float4 v = *(const float4*)(A + (size_t)(mBase + row) * F_ + k0 + c4);
      __bf16* d = As + row * 40 + c4;
      d[0] = (__bf16)v.x; d[1] = (__bf16)v.y; d[2] = (__bf16)v.z; d[3] = (__bf16)v.w;
    }
    // Stage B transposed: W[k0+kk][nBase+n] -> Bt[n][kk]
#pragma unroll
    for (int i = 0; i < 8; ++i) {
      int idx = i * 256 + t;
      int kk = idx >> 6;
      int n = idx & 63;
      Bt[n * 40 + kk] = (__bf16)W[(size_t)(k0 + kk) * HD + nBase + n];
    }
    __syncthreads();

    v16bf af0 = frag_a(As, 40, wave * 32, 0, lane);
    v16bf af1 = frag_a(As, 40, wave * 32 + 16, 0, lane);
#pragma unroll
    for (int nt = 0; nt < 4; ++nt) {
      v16bf bfr = frag_bt(Bt, 40, nt * 16, 0, lane);
      acc[0][nt] = wmma_bf16(af0, bfr, acc[0][nt]);
      acc[1][nt] = wmma_bf16(af1, bfr, acc[1][nt]);
    }
  }

  const int lanelow = (int)(lane & 15u);
  const int hiH = (int)(lane >> 4);
#pragma unroll
  for (int mt = 0; mt < 2; ++mt) {
#pragma unroll
    for (int nt = 0; nt < 4; ++nt) {
      const int n = nBase + nt * 16 + lanelow;
      const float bn = bias[n];
      const int h = n >> 6, d = n & 63;
#pragma unroll
      for (int j = 0; j < 8; ++j) {
        const int m = mBase + wave * 32 + mt * 16 + j + 8 * hiH;
        const int b = m >> 10, l = m & 1023;
        const float val = acc[mt][nt][j] + bn;
        if (z == 0)      Qo[(((size_t)b * H_ + h) * L_ + l) * D_ + d] = (__bf16)val;
        else if (z == 1) Ko[(((size_t)b * H_ + h) * L_ + l) * D_ + d] = (__bf16)val;
        else             Vo[(((size_t)b * H_ + h) * D_ + d) * L_ + l] = (__bf16)val;
      }
    }
  }
}

// ---------------------------------------------------------------------------
// Kernel 2: fused flash attention with Toeplitz bias.
// grid=(L/128, H, B), block=256. Each wave owns 16 query rows.
// K/V tiles staged via async global->LDS; next tile prefetched into L2.
// ---------------------------------------------------------------------------
__global__ __launch_bounds__(256)
void attn_kernel(const __bf16* __restrict__ Q, const __bf16* __restrict__ K,
                 const __bf16* __restrict__ Vt, const float* __restrict__ toeplitz,
                 __bf16* __restrict__ X) {
  __shared__ float  toep[4096];        // toeplitz[h], 16 KB
  __shared__ __bf16 Kb[64 * 72];       // 64 keys x 64 d (stride 72)
  __shared__ __bf16 Vb[64 * 72];       // 64 d x 64 keys (already transposed)
  __shared__ __bf16 Pb[8 * 16 * 72];   // per-wave probability tiles

  const int t = threadIdx.x;
  const unsigned lane = t & 31u;
  const int wave = t >> 5;
  const int lanelow = (int)(lane & 15u);
  const int hiH = (int)(lane >> 4);
  const int h = blockIdx.y;
  const int b = blockIdx.z;
  const int qBase = blockIdx.x * 128 + wave * 16;
  const size_t bh = (size_t)b * H_ + h;

#pragma unroll
  for (int i = 0; i < 16; ++i) toep[t + 256 * i] = toeplitz[(size_t)h * 4096 + t + 256 * i];

  // Q fragments held in registers for the whole key loop (D=64 -> 2 frags).
  const int qrow = qBase + lanelow;
  const __bf16* qp = Q + (bh * L_ + qrow) * D_ + (hiH ? 8 : 0);
  v16bf qf0 = frag_ld2(qp, 16);
  v16bf qf1 = frag_ld2(qp + 32, 16);

  float mrow[8], lrow[8];
  v8f O[4];
#pragma unroll
  for (int j = 0; j < 8; ++j) { mrow[j] = -3.0e38f; lrow[j] = 0.f; }
#pragma unroll
  for (int nt = 0; nt < 4; ++nt)
#pragma unroll
    for (int j = 0; j < 8; ++j) O[nt][j] = 0.f;

  __bf16* Pw = Pb + wave * 16 * 72;

  // Per-thread staging coordinates: 32B of one K row + 32B of one Vt row.
  const int srow = t >> 2;
  const int scol = (t & 3) * 16;
  const __bf16* kg = K + (bh * L_ + srow) * D_ + scol;   // advance by kb*D_ rows
  const __bf16* vg = Vt + (bh * D_ + srow) * L_ + scol;  // advance by kb cols

  for (int kb = 0; kb < L_; kb += 64) {
    __syncthreads();
    {
      const __bf16* ks = kg + (size_t)kb * D_;
      async_cp16(ks, Kb + srow * 72 + scol);
      async_cp16(ks + 8, Kb + srow * 72 + scol + 8);
      const __bf16* vs = vg + kb;
      async_cp16(vs, Vb + srow * 72 + scol);
      async_cp16(vs + 8, Vb + srow * 72 + scol + 8);
    }
    wait_async();
    __syncthreads();

    // Prefetch next key block into L2 while we compute on this one.
    if (kb + 64 < L_) {
      __builtin_prefetch(kg + (size_t)(kb + 64) * D_, 0, 1);
      __builtin_prefetch(vg + kb + 64, 0, 1);
    }

    // S = Q K^T  (16 queries x 64 keys; k-dim = D = 2 x 32)
    v8f S[4];
#pragma unroll
    for (int nt = 0; nt < 4; ++nt) {
#pragma unroll
      for (int j = 0; j < 8; ++j) S[nt][j] = 0.f;
      v16bf kf0 = frag_bt(Kb, 72, nt * 16, 0, lane);
      S[nt] = wmma_bf16(qf0, kf0, S[nt]);
      v16bf kf1 = frag_bt(Kb, 72, nt * 16, 32, lane);
      S[nt] = wmma_bf16(qf1, kf1, S[nt]);
    }

    // scale + Toeplitz relative-position bias (computed from C-matrix layout)
#pragma unroll
    for (int nt = 0; nt < 4; ++nt) {
      const int kpos = kb + nt * 16 + lanelow;
      const int xk = kpos >> 5, yk = kpos & 31;
#pragma unroll
      for (int j = 0; j < 8; ++j) {
        const int qpos = qBase + j + 8 * hiH;
        const int xq = qpos >> 5, yq = qpos & 31;
        const int idx = (xq - xk + 32) * 64 + (yq - yk + 32);
        S[nt][j] = S[nt][j] * 0.125f + toep[idx];
      }
    }

    // Online softmax (row reductions via shfl_xor within the 16-lane half)
#pragma unroll
    for (int j = 0; j < 8; ++j) {
      float v = fmaxf(fmaxf(S[0][j], S[1][j]), fmaxf(S[2][j], S[3][j]));
#pragma unroll
      for (int off = 1; off <= 8; off <<= 1) v = fmaxf(v, __shfl_xor(v, off, 32));
      const float mnew = fmaxf(mrow[j], v);
      const float alpha = __expf(mrow[j] - mnew);
      mrow[j] = mnew;
      float ps = 0.f;
#pragma unroll
      for (int nt = 0; nt < 4; ++nt) {
        const float p = __expf(S[nt][j] - mnew);
        S[nt][j] = p;
        ps += p;
      }
#pragma unroll
      for (int off = 1; off <= 8; off <<= 1) ps += __shfl_xor(ps, off, 32);
      lrow[j] = lrow[j] * alpha + ps;
#pragma unroll
      for (int nt = 0; nt < 4; ++nt) O[nt][j] *= alpha;
    }

    // Probabilities -> per-wave LDS tile (C layout -> A-frag layout round-trip)
#pragma unroll
    for (int nt = 0; nt < 4; ++nt)
#pragma unroll
      for (int j = 0; j < 8; ++j)
        Pw[(j + 8 * hiH) * 72 + nt * 16 + lanelow] = (__bf16)S[nt][j];
    asm volatile("s_wait_dscnt 0" ::: "memory");  // wave-internal LDS RAW fence

    // O += P V  (k-dim = 64 keys = 2 x 32)
#pragma unroll
    for (int ks = 0; ks < 64; ks += 32) {
      v16bf pf = frag_a(Pw, 72, 0, ks, lane);
#pragma unroll
      for (int nt = 0; nt < 4; ++nt) {
        v16bf vf = frag_bt(Vb, 72, nt * 16, ks, lane);
        O[nt] = wmma_bf16(pf, vf, O[nt]);
      }
    }
  }

  // Normalize and store to X[b][l][h*64+d] (bf16) for the output projection.
#pragma unroll
  for (int j = 0; j < 8; ++j) {
    const float inv = 1.f / lrow[j];
    const int qpos = qBase + j + 8 * hiH;
#pragma unroll
    for (int nt = 0; nt < 4; ++nt) {
      const int d = nt * 16 + lanelow;
      X[((size_t)b * L_ + qpos) * HD + (size_t)h * D_ + d] = (__bf16)(O[nt][j] * inv);
    }
  }
}

// ---------------------------------------------------------------------------
// Kernel 3: output projection X(bf16) @ Wo + bo -> f32. grid=(64, 8), block=256.
// A tile staged via async global->LDS (pure bf16 copy).
// ---------------------------------------------------------------------------
__global__ __launch_bounds__(256)
void out_gemm(const __bf16* __restrict__ X, const float* __restrict__ Wo,
              const float* __restrict__ bo, float* __restrict__ out) {
  __shared__ __bf16 As[256 * 40];
  __shared__ __bf16 Bt[64 * 40];

  const int t = threadIdx.x;
  const unsigned lane = t & 31u;
  const int wave = t >> 5;
  const int mBase = blockIdx.x * 256;
  const int nBase = blockIdx.y * 64;

  v8f acc[2][4];
#pragma unroll
  for (int mt = 0; mt < 2; ++mt)
#pragma unroll
    for (int nt = 0; nt < 4; ++nt)
#pragma unroll
      for (int j = 0; j < 8; ++j) acc[mt][nt][j] = 0.f;

  for (int k0 = 0; k0 < HD; k0 += 32) {
    __syncthreads();
    // Stage A: 256x32 bf16 (1024 16B chunks, 4 per thread) via ASYNCcnt DMA
#pragma unroll
    for (int i = 0; i < 4; ++i) {
      int idx = i * 256 + t;
      int row = idx >> 2;
      int c = (idx & 3) * 8;
      async_cp16(X + (size_t)(mBase + row) * HD + k0 + c, As + row * 40 + c);
    }
    // Stage B transposed (f32 -> bf16 conversion, VALU path)
#pragma unroll
    for (int i = 0; i < 8; ++i) {
      int idx = i * 256 + t;
      int kk = idx >> 6;
      int n = idx & 63;
      Bt[n * 40 + kk] = (__bf16)Wo[(size_t)(k0 + kk) * F_ + nBase + n];
    }
    wait_async();
    __syncthreads();

    v16bf af0 = frag_a(As, 40, wave * 32, 0, lane);
    v16bf af1 = frag_a(As, 40, wave * 32 + 16, 0, lane);
#pragma unroll
    for (int nt = 0; nt < 4; ++nt) {
      v16bf bfr = frag_bt(Bt, 40, nt * 16, 0, lane);
      acc[0][nt] = wmma_bf16(af0, bfr, acc[0][nt]);
      acc[1][nt] = wmma_bf16(af1, bfr, acc[1][nt]);
    }
  }

  const int lanelow = (int)(lane & 15u);
  const int hiH = (int)(lane >> 4);
#pragma unroll
  for (int mt = 0; mt < 2; ++mt) {
#pragma unroll
    for (int nt = 0; nt < 4; ++nt) {
      const int n = nBase + nt * 16 + lanelow;
      const float bn = bo[n];
#pragma unroll
      for (int j = 0; j < 8; ++j) {
        const int m = mBase + wave * 32 + mt * 16 + j + 8 * hiH;
        out[(size_t)m * F_ + n] = acc[mt][nt][j] + bn;
      }
    }
  }
}

// ---------------------------------------------------------------------------
extern "C" void kernel_launch(void* const* d_in, const int* in_sizes, int n_in,
                              void* d_out, int out_size, void* d_ws, size_t ws_size,
                              hipStream_t stream) {
  const float* inq      = (const float*)d_in[0];
  const float* inkv     = (const float*)d_in[1];
  const float* Wq       = (const float*)d_in[2];
  const float* bq       = (const float*)d_in[3];
  const float* Wk       = (const float*)d_in[4];
  const float* bk       = (const float*)d_in[5];
  const float* Wv       = (const float*)d_in[6];
  const float* bv       = (const float*)d_in[7];
  const float* Wo       = (const float*)d_in[8];
  const float* bo       = (const float*)d_in[9];
  const float* toeplitz = (const float*)d_in[10];

  __bf16* ws = (__bf16*)d_ws;
  const size_t QKVN = (size_t)B_ * L_ * HD;  // 8,388,608 elements (16 MB bf16)
  __bf16* Qo = ws;
  __bf16* Ko = ws + QKVN;
  __bf16* Vo = ws + 2 * QKVN;
  __bf16* Xo = ws + 3 * QKVN;

  qkv_gemm<<<dim3(64, 8, 3), 256, 0, stream>>>(inq, inkv, Wq, bq, Wk, bk, Wv, bv,
                                               Qo, Ko, Vo);
  attn_kernel<<<dim3(L_ / 128, H_, B_), 256, 0, stream>>>(Qo, Ko, Vo, toeplitz, Xo);
  out_gemm<<<dim3(64, 8), 256, 0, stream>>>(Xo, Wo, bo, (float*)d_out);
}
